// Cam2BEV_55027120996694
// MI455X (gfx1250) — compile-verified, matrix operand order checked by reading, not run on hardware
//
#include <hip/hip_runtime.h>
#include <hip/hip_bf16.h>

typedef __attribute__((ext_vector_type(16))) _Float16 v16h;
typedef __attribute__((ext_vector_type(8)))  _Float16 v8h;
typedef __attribute__((ext_vector_type(8)))  float    v8f;
typedef __attribute__((ext_vector_type(4)))  unsigned int v4u;
typedef __attribute__((ext_vector_type(8)))  int v8i;
typedef __attribute__((ext_vector_type(4)))  int v4i;

#define NCAMS 6
#define FCH   128
#define FHH   32
#define FWW   88
#define HWC   (FHH*FWW)          // 2816
#define NPIX  (NCAMS*HWC)        // 16896
#define NDEP  41
#define OUTC  128
#define C3    170                // NDEP + 1 + OUTC
#define C3P   176                // padded
#define BEVW  100
#define BEVP  10000
#define NG    NPIX               // 16896 gaussians
#define GK    64                 // gaussians composited per sync round

#ifdef __has_builtin
#if __has_builtin(__builtin_amdgcn_tensor_load_to_lds)
#define HAVE_TDM 1
#endif
#endif
#ifndef HAVE_TDM
#define HAVE_TDM 0
#endif

union V16 { v16h v; v8h h[2]; };

__device__ __forceinline__ v8h ld8h(const _Float16* p) {
  return *reinterpret_cast<const v8h*>(p);
}
__device__ __forceinline__ v8h zero8() {
  v8h z;
#pragma unroll
  for (int i = 0; i < 8; ++i) z[i] = (_Float16)0.f;
  return z;
}

__device__ __forceinline__ void inv3(const float* A, float* o) {
  float a=A[0],b=A[1],c=A[2],d=A[3],e=A[4],f=A[5],g=A[6],h=A[7],i=A[8];
  float C0 = e*i - f*h, C1 = f*g - d*i, C2 = d*h - e*g;
  float id = 1.0f / (a*C0 + b*C1 + c*C2);
  o[0]=C0*id;        o[1]=(c*h-b*i)*id; o[2]=(b*f-c*e)*id;
  o[3]=C1*id;        o[4]=(a*i-c*g)*id; o[5]=(c*d-a*f)*id;
  o[6]=C2*id;        o[7]=(b*g-a*h)*id; o[8]=(a*e-b*d)*id;
}

#if HAVE_TDM
// TDM descriptor-driven copy: 2048B contiguous (512 dwords) global -> LDS.
// D# group0: count=1, lds_addr, 57b global addr, type=2 ("image").
// D# group1: data_size=4B (code 2), tile_dim0=512, tile_dim1=1, huge tensor_dim0.
__device__ __forceinline__ void tdm_load_2k(const float* src, unsigned lds_off) {
  unsigned long long ga = (unsigned long long)(uintptr_t)src;
  v4u g0v;
  g0v.x = 1u;                                            // count=1, user mode
  g0v.y = lds_off;                                       // lds_addr
  g0v.z = (unsigned)(ga & 0xffffffffull);                // global_addr[31:0]
  g0v.w = (unsigned)((ga >> 32) & 0x01ffffffull) | 0x80000000u; // [56:32] | type=2
  const unsigned dim0 = 0x10000000u;                     // big tensor_dim0: no OOB clamp
  v8i g1v;
  g1v[0] = (int)(2u << 16);                              // data_size=4B
  g1v[1] = (int)((dim0 & 0xffffu) << 16);                // tensor_dim0[15:0]
  g1v[2] = (int)((dim0 >> 16) & 0xffffu);                // tensor_dim0[31:16], dim1=0
  g1v[3] = (int)(512u << 16);                            // tile_dim0 = 512 dwords
  g1v[4] = 1;                                            // tile_dim1 = 1
  g1v[5] = (int)dim0;                                    // dim0_stride[31:0]
  g1v[6] = 0;
  g1v[7] = 0;
  v4i z4 = {0,0,0,0};
#if __clang_major__ >= 23
  v8i z8 = {0,0,0,0,0,0,0,0};
  __builtin_amdgcn_tensor_load_to_lds(g0v, g1v, z4, z4, z8, 0);
#else
  __builtin_amdgcn_tensor_load_to_lds(g0v, g1v, z4, z4, 0);
#endif
}
#endif

// ---------------- prep: camera matrices, folded BN consts, counter ----------------
__global__ void __launch_bounds__(256) k_prep(
    const float* __restrict__ rots, const float* __restrict__ trans,
    const float* __restrict__ intr, const float* __restrict__ prots,
    const float* __restrict__ ptrans,
    const float* __restrict__ c1b, const float* __restrict__ g1,
    const float* __restrict__ b1,  const float* __restrict__ m1,
    const float* __restrict__ v1,
    const float* __restrict__ c2b, const float* __restrict__ g2,
    const float* __restrict__ b2,  const float* __restrict__ m2,
    const float* __restrict__ v2,
    const float* __restrict__ c3b,
    float* __restrict__ cams, float* __restrict__ ab1,
    float* __restrict__ ab2, float* __restrict__ b3, float* __restrict__ cnt)
{
  int t = threadIdx.x;
  if (t < NCAMS) {
    float ipr[9], ik[9], comb[9];
    inv3(prots + t*9, ipr);
    inv3(intr  + t*9, ik);
    const float* R = rots + t*9;
#pragma unroll
    for (int i = 0; i < 3; ++i)
#pragma unroll
      for (int j = 0; j < 3; ++j)
        comb[i*3+j] = R[i*3+0]*ik[0+j] + R[i*3+1]*ik[3+j] + R[i*3+2]*ik[6+j];
    float* cm = cams + t*24;
#pragma unroll
    for (int i = 0; i < 9; ++i) cm[i]    = ipr[i];
#pragma unroll
    for (int i = 0; i < 3; ++i) cm[9+i]  = ptrans[t*3+i];
#pragma unroll
    for (int i = 0; i < 9; ++i) cm[12+i] = comb[i];
#pragma unroll
    for (int i = 0; i < 3; ++i) cm[21+i] = trans[t*3+i];
  }
  if (t == 6) *cnt = 0.0f;
  if (t < FCH) {
    float s1 = g1[t] * rsqrtf(v1[t] + 1e-3f);
    ab1[t] = s1; ab1[FCH+t] = (c1b[t]-m1[t])*s1 + b1[t];
    float s2 = g2[t] * rsqrtf(v2[t] + 1e-3f);
    ab2[t] = s2; ab2[FCH+t] = (c2b[t]-m2[t])*s2 + b2[t];
  }
  if (t < C3P) b3[t] = (t < C3) ? c3b[t] : 0.0f;
}

// ---------------- weight repack to f16 WMMA-A layout [tap][out][in] ----------------
__global__ void __launch_bounds__(256) k_wconv(
    const float* __restrict__ w1, const float* __restrict__ w2,
    const float* __restrict__ w3,
    _Float16* __restrict__ wA1, _Float16* __restrict__ wA2,
    _Float16* __restrict__ wA3)
{
  const int NW = 9*FCH*FCH;      // 147456
  int idx = blockIdx.x*256 + threadIdx.x;
  if (idx < NW) {
    int i = idx & 127; int o = (idx >> 7) & 127; int tap = idx >> 14;
    wA1[idx] = (_Float16)w1[(size_t)(o*FCH + i)*9 + tap];
  } else if (idx < 2*NW) {
    int j = idx - NW;
    int i = j & 127; int o = (j >> 7) & 127; int tap = j >> 14;
    wA2[j] = (_Float16)w2[(size_t)(o*FCH + i)*9 + tap];
  } else if (idx < 2*NW + C3P*FCH) {
    int j = idx - 2*NW;
    int i = j & 127; int o = j >> 7;
    wA3[j] = (o < C3) ? (_Float16)w3[(size_t)o*FCH + i] : (_Float16)0.f;
  }
}

// ---------------- transpose img_feats -> channel-last f16 ----------------
__global__ void __launch_bounds__(256) k_x0(
    const float* __restrict__ img, _Float16* __restrict__ x0)
{
  int idx = blockIdx.x*256 + threadIdx.x;   // NPIX*128 exact
  int c = idx & 127; int pix = idx >> 7;
  int cam = pix / HWC; int hw = pix - cam*HWC;
  x0[idx] = (_Float16)img[(size_t)(cam*FCH + c)*HWC + hw];
}

// ---------------- 3x3 conv + folded BN + ReLU, WMMA implicit GEMM ----------------
__global__ void __launch_bounds__(256) k_conv3x3(
    const _Float16* __restrict__ xin,   // [NPIX][128] channel-last
    const _Float16* __restrict__ wA,    // [9][128][128]
    const float*    __restrict__ ab,    // [0..127]=scale [128..255]=bias
    _Float16* __restrict__ xout)        // [NPIX][128]
{
  int tile = blockIdx.x;
  int wave = threadIdx.x >> 5;
  int lane = threadIdx.x & 31;
  int ln = lane & 15, khalf = lane >> 4;
  int pix = tile*16 + ln;
  int cam = pix / HWC; int hw = pix - cam*HWC;
  int h = hw / FWW, w = hw - (hw/FWW)*FWW;
  int mbase = wave*16;
  v8f acc = {0.f,0.f,0.f,0.f,0.f,0.f,0.f,0.f};

  for (int tap = 0; tap < 9; ++tap) {
    int dy = tap/3 - 1, dx = tap - (tap/3)*3 - 1;
    int hh = h + dy, ww = w + dx;
    bool inb = (hh >= 0) & (hh < FHH) & (ww >= 0) & (ww < FWW);
    int hc = min(max(hh,0),FHH-1), wc = min(max(ww,0),FWW-1);
    const _Float16* bsrc = xin + ((size_t)cam*HWC + hc*FWW + wc)*FCH;
    const _Float16* asrc = wA + ((size_t)tap*FCH + mbase + ln)*FCH;
    if (tap < 8) {  // WGP-scope prefetch of next tap's activation row
      int dy2 = (tap+1)/3 - 1, dx2 = (tap+1) - ((tap+1)/3)*3 - 1;
      int h2 = min(max(h+dy2,0),FHH-1), w2 = min(max(w+dx2,0),FWW-1);
      __builtin_prefetch(xin + ((size_t)cam*HWC + h2*FWW + w2)*FCH, 0, 3);
    }
#pragma unroll
    for (int kc = 0; kc < FCH; kc += 32) {
      V16 a, b;
      a.h[0] = ld8h(asrc + kc +      khalf*8);
      a.h[1] = ld8h(asrc + kc + 16 + khalf*8);
      b.h[0] = ld8h(bsrc + kc +      khalf*8);
      b.h[1] = ld8h(bsrc + kc + 16 + khalf*8);
      if (!inb) { b.h[0] = zero8(); b.h[1] = zero8(); }
      acc = __builtin_amdgcn_wmma_f32_16x16x32_f16(
          false, a.v, false, b.v, (short)0, acc, false, false);
    }
  }
  int cb = mbase + khalf*8;
  v8h ov;
#pragma unroll
  for (int r = 0; r < 8; ++r) {
    int c = cb + r;
    float y = acc[r]*ab[c] + ab[FCH+c];
    ov[r] = (_Float16)(y > 0.f ? y : 0.f);
  }
  *reinterpret_cast<v8h*>(xout + (size_t)pix*FCH + cb) = ov;
}

// ---------------- 1x1 conv 128->170 (f32 out), 11 waves ----------------
__global__ void __launch_bounds__(352) k_conv1x1(
    const _Float16* __restrict__ xin,   // [NPIX][128]
    const _Float16* __restrict__ wA,    // [176][128]
    const float*    __restrict__ b3,    // [176]
    float* __restrict__ xout)           // [NPIX][176]
{
  int tile = blockIdx.x;
  int wave = threadIdx.x >> 5;          // 0..10
  int lane = threadIdx.x & 31;
  int ln = lane & 15, khalf = lane >> 4;
  int pix = tile*16 + ln;
  int mbase = wave*16;
  v8f acc = {0.f,0.f,0.f,0.f,0.f,0.f,0.f,0.f};
  const _Float16* bsrc = xin + (size_t)pix*FCH;
  const _Float16* asrc = wA + (size_t)(mbase + ln)*FCH;
#pragma unroll
  for (int kc = 0; kc < FCH; kc += 32) {
    V16 a, b;
    a.h[0] = ld8h(asrc + kc +      khalf*8);
    a.h[1] = ld8h(asrc + kc + 16 + khalf*8);
    b.h[0] = ld8h(bsrc + kc +      khalf*8);
    b.h[1] = ld8h(bsrc + kc + 16 + khalf*8);
    acc = __builtin_amdgcn_wmma_f32_16x16x32_f16(
        false, a.v, false, b.v, (short)0, acc, false, false);
  }
  int cb = mbase + khalf*8;
#pragma unroll
  for (int r = 0; r < 8; ++r) {
    int c = cb + r;
    if (c < C3) xout[(size_t)pix*C3P + c] = acc[r] + b3[c];
  }
}

// ---------------- per-gaussian moments: softmax depth, geometry, conic ----------------
__global__ void __launch_bounds__(256) k_moments(
    const float* __restrict__ x3,       // [NPIX][176]
    const float* __restrict__ cams,     // [6][24]: IPR,PT,COMB,T
    float* __restrict__ gparams,        // [NG][8]: my,mx,cA,cB,cC,op
    _Float16* __restrict__ colors_t,    // [128][NG] channel-major
    float* __restrict__ cnt)
{
  int g = blockIdx.x*256 + threadIdx.x;
  if (g >= NG) return;
  int cam = g / HWC; int hw = g - cam*HWC;
  int h = hw / FWW, w = hw - (hw/FWW)*FWW;
  const float* cm  = cams + cam*24;
  const float* row = x3 + (size_t)g*C3P;

  float mxl = -1e30f;
  for (int d = 0; d < NDEP; ++d) mxl = fmaxf(mxl, row[d]);
  float ssum = 0.f;
  for (int d = 0; d < NDEP; ++d) ssum += __expf(row[d]-mxl);
  float inv = 1.f/ssum;

  float xs = (float)w * (703.0f/87.0f);
  float ys = (float)h * (255.0f/31.0f);
  float s1x=0.f, s1y=0.f, sxx=0.f, sxy=0.f, syy=0.f;
  for (int d = 0; d < NDEP; ++d) {
    float pw = __expf(row[d]-mxl) * inv;
    float ds = 4.0f + (float)d;
    float q0 = xs - cm[9], q1 = ys - cm[10], q2 = ds - cm[11];
    float r0 = cm[0]*q0 + cm[1]*q1 + cm[2]*q2;
    float r1 = cm[3]*q0 + cm[4]*q1 + cm[5]*q2;
    float r2 = cm[6]*q0 + cm[7]*q1 + cm[8]*q2;
    float u0 = r0*r2, u1 = r1*r2, u2 = r2;
    float ex = cm[12]*u0 + cm[13]*u1 + cm[14]*u2 + cm[21];
    float ey = cm[15]*u0 + cm[16]*u1 + cm[17]*u2 + cm[22];
    s1x += pw*ex; s1y += pw*ey;
    sxx += pw*ex*ex; sxy += pw*ex*ey; syy += pw*ey*ey;
  }
  const float k9 = 1.0f/9.0f;
  float cxx = (sxx - s1x*s1x)*k9*4e-4f;
  float cxy = (sxy - s1x*s1y)*k9*4e-4f;
  float cyy = (syy - s1y*s1y)*k9*4e-4f;
  float mnx = s1x*0.02f, mny = s1y*0.02f;
  float a  = 2500.f*cyy + 0.3f;
  float c2 = 2500.f*cxx + 0.3f;
  float b2 = 2500.f*cxy;
  float det = a*c2 - b2*b2;
  float op  = 1.0f/(1.0f + __expf(-row[NDEP]));
  bool  mask  = (op > 0.05f);
  bool  valid = mask && (det > 0.f);
  float idet = valid ? 1.f/det : 0.f;
  float* gp = gparams + (size_t)g*8;
  gp[0] = -50.f*mny + 50.f;          // my (pixel row)
  gp[1] = -50.f*mnx + 50.f;          // mx (pixel col)
  gp[2] =  c2*idet;                  // conic A
  gp[3] = -b2*idet;                  // conic B
  gp[4] =  a*idet;                   // conic C
  gp[5] = valid ? op : 0.f;
  gp[6] = 0.f; gp[7] = 0.f;
  for (int c = 0; c < OUTC; ++c)
    colors_t[(size_t)c*NG + g] = (_Float16)row[NDEP+1+c];
  if (mask) atomicAdd(cnt, 1.0f);
}

// ---------------- ordered alpha-composite splat with WMMA accumulation ----------------
// block = 8 waves, one 16-pixel tile, all 128 channels, 264 rounds of 64 gaussians.
// TDM double-buffers the gaussian-parameter records into LDS under TENSORcnt.
__global__ void __launch_bounds__(256) k_splat(
    const float* __restrict__ gparams,
    const _Float16* __restrict__ colors_t,   // [128][NG]
    float* __restrict__ bev)                 // [128][100][100]
{
  __shared__ alignas(16) float eLDS[GK*16];
  __shared__ alignas(16) _Float16 wLDS[16*GK];   // [pixel][k]
  __shared__ float TLDS[16];
#if HAVE_TDM
  __shared__ alignas(16) float gbuf[2][GK*8];    // double-buffered gaussian params
#endif

  int tid  = threadIdx.x;
  int lane = tid & 31;
  int wave = tid >> 5;
  int ln = lane & 15, khalf = lane >> 4;
  int P0 = blockIdx.x * 16;
  int mbase = wave*16;

  int ep = tid & 15;           // pixel this thread evaluates alphas for
  int eg = tid >> 4;           // base gaussian slot (does eg, eg+16, eg+32, eg+48)
  float pi_e = (float)((P0 + ep) / BEVW);
  float pj_e = (float)((P0 + ep) % BEVW);
  if (tid < 16) TLDS[tid] = 1.0f;
#if HAVE_TDM
  if (wave == 0) {
    tdm_load_2k(gparams, (unsigned)(uintptr_t)&gbuf[0][0]);
    __builtin_amdgcn_s_wait_tensorcnt(0);
  }
#endif
  __syncthreads();

  v8f acc = {0.f,0.f,0.f,0.f,0.f,0.f,0.f,0.f};
  const _Float16* asrc = colors_t + (size_t)(mbase + ln)*NG;
  const int NCH = NG / GK;     // 264

  for (int ic = 0; ic < NCH; ++ic) {
    int g0 = ic * GK;
#if HAVE_TDM
    if (wave == 0 && ic + 1 < NCH)
      tdm_load_2k(gparams + (size_t)(g0 + GK)*8,
                  (unsigned)(uintptr_t)&gbuf[(ic+1)&1][0]);
    const float* gsrc = &gbuf[ic&1][0];
#else
    const float* gsrc = gparams + (size_t)g0*8;
#endif
    // (a) 1024 alpha evaluations spread over 256 threads (4 each)
#pragma unroll
    for (int gg = 0; gg < 4; ++gg) {
      int k = eg + gg*16;
      const float* gp = gsrc + k*8;
      float dI = pi_e - gp[0], dJ = pj_e - gp[1];
      float q  = gp[2]*dI*dI + 2.f*gp[3]*dI*dJ + gp[4]*dJ*dJ;
      float al = fminf(gp[5]*__expf(-0.5f*q), 0.99f);
      if (!(q >= 0.f) || al < (1.0f/255.0f)) al = 0.f;
      eLDS[k*16 + ep] = al;
    }
    __syncthreads();
    // (b) serial transmittance recurrence: one lane per pixel
    if (tid < 16) {
      float T = TLDS[tid];
#pragma unroll
      for (int k = 0; k < GK; ++k) {
        float al = eLDS[k*16 + tid];
        wLDS[tid*GK + k] = (_Float16)(al * T);
        T *= (1.0f - al);
      }
      TLDS[tid] = T;
    }
    __syncthreads();
    // WGP-scope prefetch of next colors A-tile while WMMA runs
    __builtin_prefetch(asrc + g0 + GK, 0, 3);
    // (c) acc[16 chan x 16 pix] += colors^T[16x32] @ w[32x16], twice
#pragma unroll
    for (int ks = 0; ks < 2; ++ks) {
      V16 a, b;
      a.h[0] = ld8h(asrc + g0 + ks*32 +      khalf*8);
      a.h[1] = ld8h(asrc + g0 + ks*32 + 16 + khalf*8);
      b.h[0] = *reinterpret_cast<const v8h*>(&wLDS[ln*GK + ks*32 +      khalf*8]);
      b.h[1] = *reinterpret_cast<const v8h*>(&wLDS[ln*GK + ks*32 + 16 + khalf*8]);
      acc = __builtin_amdgcn_wmma_f32_16x16x32_f16(
          false, a.v, false, b.v, (short)0, acc, false, false);
    }
#if HAVE_TDM
    if (wave == 0) __builtin_amdgcn_s_wait_tensorcnt(0);
#endif
    __syncthreads();
  }
  int P  = P0 + ln;
  int cb = mbase + khalf*8;
#pragma unroll
  for (int r = 0; r < 8; ++r)
    bev[(size_t)(cb + r)*BEVP + P] = acc[r];
}

// ---------------- launch ----------------
extern "C" void kernel_launch(void* const* d_in, const int* in_sizes, int n_in,
                              void* d_out, int out_size, void* d_ws, size_t ws_size,
                              hipStream_t stream) {
  (void)in_sizes; (void)n_in; (void)out_size; (void)ws_size;
  const float* rots   = (const float*)d_in[0];
  const float* trans  = (const float*)d_in[1];
  const float* intr   = (const float*)d_in[2];
  const float* prots  = (const float*)d_in[3];
  const float* ptrans = (const float*)d_in[4];
  const float* img    = (const float*)d_in[5];
  const float* w1  = (const float*)d_in[6];  const float* c1b = (const float*)d_in[7];
  const float* g1  = (const float*)d_in[8];  const float* b1  = (const float*)d_in[9];
  const float* m1  = (const float*)d_in[10]; const float* v1  = (const float*)d_in[11];
  const float* w2  = (const float*)d_in[12]; const float* c2b = (const float*)d_in[13];
  const float* g2  = (const float*)d_in[14]; const float* b2  = (const float*)d_in[15];
  const float* m2  = (const float*)d_in[16]; const float* v2  = (const float*)d_in[17];
  const float* w3  = (const float*)d_in[18]; const float* c3b = (const float*)d_in[19];

  char* ws = (char*)d_ws;
  const size_t OFF_CAMS = 0;
  const size_t OFF_AB1  = 1024;
  const size_t OFF_AB2  = 2048;
  const size_t OFF_B3   = 3072;
  const size_t OFF_WA1  = 4096;
  const size_t OFF_WA2  = OFF_WA1 + (size_t)9*FCH*FCH*2;      //  294912 B
  const size_t OFF_WA3  = OFF_WA2 + (size_t)9*FCH*FCH*2;
  const size_t OFF_X0   = OFF_WA3 + (size_t)C3P*FCH*2;
  const size_t OFF_X1   = OFF_X0  + (size_t)NPIX*FCH*2;
  const size_t OFF_X2   = OFF_X1  + (size_t)NPIX*FCH*2;
  const size_t OFF_X3   = OFF_X2  + (size_t)NPIX*FCH*2;
  const size_t OFF_GP   = OFF_X3  + (size_t)NPIX*C3P*4;
  const size_t OFF_COL  = OFF_GP  + (size_t)NG*8*4;

  float*     cams = (float*)(ws + OFF_CAMS);
  float*     ab1  = (float*)(ws + OFF_AB1);
  float*     ab2  = (float*)(ws + OFF_AB2);
  float*     b3   = (float*)(ws + OFF_B3);
  _Float16*  wA1  = (_Float16*)(ws + OFF_WA1);
  _Float16*  wA2  = (_Float16*)(ws + OFF_WA2);
  _Float16*  wA3  = (_Float16*)(ws + OFF_WA3);
  _Float16*  x0   = (_Float16*)(ws + OFF_X0);
  _Float16*  x1   = (_Float16*)(ws + OFF_X1);
  _Float16*  x2   = (_Float16*)(ws + OFF_X2);
  float*     x3   = (float*)(ws + OFF_X3);
  float*     gpar = (float*)(ws + OFF_GP);
  _Float16*  colT = (_Float16*)(ws + OFF_COL);

  float* bev = (float*)d_out;
  float* cnt = bev + (size_t)OUTC*BEVP;   // d_out[1280000] = num_gaussians

  k_prep<<<1, 256, 0, stream>>>(rots, trans, intr, prots, ptrans,
                                c1b, g1, b1, m1, v1,
                                c2b, g2, b2, m2, v2, c3b,
                                cams, ab1, ab2, b3, cnt);
  k_wconv<<<1240, 256, 0, stream>>>(w1, w2, w3, wA1, wA2, wA3);
  k_x0<<<(NPIX*FCH)/256, 256, 0, stream>>>(img, x0);
  k_conv3x3<<<NPIX/16, 256, 0, stream>>>(x0, wA1, ab1, x1);
  k_conv3x3<<<NPIX/16, 256, 0, stream>>>(x1, wA2, ab2, x2);
  k_conv1x1<<<NPIX/16, 352, 0, stream>>>(x2, wA3, b3, x3);
  k_moments<<<(NG+255)/256, 256, 0, stream>>>(x3, cams, gpar, colT, bev + (size_t)OUTC*BEVP);
  k_splat<<<BEVP/16, 256, 0, stream>>>(gpar, colT, bev);
}
